// Attention_87342454931951
// MI455X (gfx1250) — matrix operand order repaired and checked by reading, not compiled, not run on hardware
//
#include <hip/hip_runtime.h>
#include <hip/hip_bf16.h>

// ---------------------------------------------------------------------------
// MI455X (gfx1250): bf16 WMMA + double-buffered async global->LDS staging.
// B=2,S=1024,HID=4096,H=32,HKV=8,HD=128. ~206 GFLOP, compute bound ->
// v_wmma_f32_16x16x32_bf16 with GLOBAL_LOAD_ASYNC_TO_LDS_B128 prefetch
// overlapped with compute (ASYNCcnt pipelining). All operands bf16
// (weights pre-transposed to [N,K]) so staging is pure contiguous DMA.
// ---------------------------------------------------------------------------

#define B_ 2
#define S_ 1024
#define HID_ 4096
#define H_ 32
#define HKV_ 8
#define HD_ 128
#define G_ (H_ / HKV_)   // 4

typedef __attribute__((ext_vector_type(16))) __bf16 bf16x16;
typedef __attribute__((ext_vector_type(8)))  __bf16 bf16x8;
typedef __attribute__((ext_vector_type(8)))  float  f32x8;

union FragU { bf16x16 v; bf16x8 h[2]; __bf16 e[16]; };

// Async 16B global->LDS copy; INST_OFFSET applies to BOTH addresses, so one
// base pair serves a whole row of chunks (no per-chunk 64-bit adds).
template <int OFF>
__device__ __forceinline__ void async_copy16_o(unsigned lds_off, const void* g) {
  asm volatile("global_load_async_to_lds_b128 %0, %1, off offset:%2"
               : : "v"(lds_off), "v"(g), "n"(OFF) : "memory");
}
// Copy 64 contiguous bf16 (128B) global -> LDS.
__device__ __forceinline__ void stage_row128B(unsigned lds_off, const __bf16* g) {
  async_copy16_o<0>(lds_off, g);
  async_copy16_o<16>(lds_off, g);
  async_copy16_o<32>(lds_off, g);
  async_copy16_o<48>(lds_off, g);
  async_copy16_o<64>(lds_off, g);
  async_copy16_o<80>(lds_off, g);
  async_copy16_o<96>(lds_off, g);
  async_copy16_o<112>(lds_off, g);
}
__device__ __forceinline__ void stage_row64B(unsigned lds_off, const __bf16* g) {
  async_copy16_o<0>(lds_off, g);
  async_copy16_o<16>(lds_off, g);
  async_copy16_o<32>(lds_off, g);
  async_copy16_o<48>(lds_off, g);
}
__device__ __forceinline__ void wait_async() {
  asm volatile("s_wait_asynccnt 0x0" ::: "memory");
}
__device__ __forceinline__ unsigned lds_addr(const void* p) {
  return (unsigned)(size_t)p;  // flat LDS aperture: addr[31:0] = LDS offset
}

// A-fragment (16x32 bf16, M x K): lane L holds row M=L%16,
// K = {8*(L/16)+0..7} in e[0..7] and {16+8*(L/16)+0..7} in e[8..15].
__device__ __forceinline__ bf16x16 frag_a(const __bf16* lds, int stride,
                                          int row, int k0, int lane) {
  const __bf16* p = lds + row * stride + k0 + 8 * (lane >> 4);
  FragU u;
  u.h[0] = *(const bf16x8*)(p);
  u.h[1] = *(const bf16x8*)(p + 16);
  return u.v;
}

// B-fragment (32x16 bf16, K x N) from LDS holding B^T ([n][k], contiguous k):
// lane L holds col N=L%16, K = {8*(L/16)+0..7} in e[0..7] and {16+8*(L/16)+0..7} in e[8..15].
__device__ __forceinline__ bf16x16 frag_b(const __bf16* lds, int stride,
                                          int nrow, int k0, int lane) {
  const __bf16* p = lds + nrow * stride + k0 + 8 * (lane >> 4);
  FragU u;
  u.h[0] = *(const bf16x8*)(p);
  u.h[1] = *(const bf16x8*)(p + 16);
  return u.v;
}

__device__ __forceinline__ f32x8 wmma_bf16(bf16x16 a, bf16x16 b, f32x8 c) {
  return __builtin_amdgcn_wmma_f32_16x16x32_bf16(false, a, false, b,
                                                 (short)0, c, false, false);
}

// ---------------------------------------------------------------------------
// Pre-conversion kernels (one-time, bandwidth-trivial).
// ---------------------------------------------------------------------------
__global__ void __launch_bounds__(256)
cast_f32_bf16_kernel(const float* __restrict__ src, __bf16* __restrict__ dst,
                     int n) {
  int i = blockIdx.x * 256 + threadIdx.x;
  if (i < n) dst[i] = (__bf16)src[i];
}

// src: f32 [K,N] row-major -> dst: bf16 [N,K] row-major (transposed weights).
__global__ void __launch_bounds__(256)
transpose_f32_bf16_kernel(const float* __restrict__ src, __bf16* __restrict__ dst,
                          int K, int N) {
  int i = blockIdx.x * 256 + threadIdx.x;
  if (i >= K * N) return;
  int k = i / N, n = i - k * N;
  dst[(size_t)n * K + k] = (__bf16)src[i];
}

// ---------------------------------------------------------------------------
// GEMM: C[M,N] = A[M,K] @ B[K,N]; A bf16 [M,K], B passed transposed bf16
// [N,K]. Block tile 128x128, 256 threads / 8 waves, wave tile 32x64.
// Double-buffered LDS; next K-tile prefetched by async DMA during compute.
// ---------------------------------------------------------------------------
template <typename TOUT>
__global__ void __launch_bounds__(256)
gemm_bf16_async_kernel(const __bf16* __restrict__ A, const __bf16* __restrict__ Bt,
                       TOUT* __restrict__ C, int M, int N, int K) {
  __shared__ __attribute__((aligned(16))) __bf16 As[2][128 * 64]; // [row][k]
  __shared__ __attribute__((aligned(16))) __bf16 Bs[2][128 * 64]; // [n][k]

  const int tid  = threadIdx.x;
  const int lane = tid & 31;
  const int wave = tid >> 5;
  const int wm = wave & 3;   // wave rows wm*32..+31
  const int wn = wave >> 2;  // wave cols wn*64..+63
  const int mBase = blockIdx.y * 128;
  const int nBase = blockIdx.x * 128;

  f32x8 acc[2][4];
#pragma unroll
  for (int mi = 0; mi < 2; ++mi)
#pragma unroll
    for (int ni = 0; ni < 4; ++ni) acc[mi][ni] = (f32x8)0.0f;

  // Staging map: thread t copies 64B of row t/2 (half (t&1)) of each tile.
  const int srow = tid >> 1;
  const int scb  = (tid & 1) * 32;  // element base within 64
  const unsigned laA[2] = {lds_addr(As[0] + srow * 64 + scb),
                           lds_addr(As[1] + srow * 64 + scb)};
  const unsigned laB[2] = {lds_addr(Bs[0] + srow * 64 + scb),
                           lds_addr(Bs[1] + srow * 64 + scb)};
  const __bf16* gA = A  + (size_t)(mBase + srow) * K + scb;
  const __bf16* gB = Bt + (size_t)(nBase + srow) * K + scb;

  // Prologue: stage first tile into buffer 0.
  stage_row64B(laA[0], gA);
  stage_row64B(laB[0], gB);

  int cur = 0;
  for (int kb = 0; kb < K; kb += 64, cur ^= 1) {
    wait_async();      // own prefetch done
    __syncthreads();   // everyone's prefetch done; 'cur^1' readers done too
    if (kb + 64 < K) { // prefetch next tile while computing this one
      stage_row64B(laA[cur ^ 1], gA + kb + 64);
      stage_row64B(laB[cur ^ 1], gB + kb + 64);
    }
    const __bf16* as = As[cur];
    const __bf16* bs = Bs[cur];
#pragma unroll
    for (int kk = 0; kk < 64; kk += 32) {
      bf16x16 a0 = frag_a(as, 64, wm * 32 + (lane & 15), kk, lane);
      bf16x16 a1 = frag_a(as, 64, wm * 32 + 16 + (lane & 15), kk, lane);
      bf16x16 bfr[4];
#pragma unroll
      for (int ni = 0; ni < 4; ++ni)
        bfr[ni] = frag_b(bs, 64, wn * 64 + ni * 16 + (lane & 15), kk, lane);
#pragma unroll
      for (int ni = 0; ni < 4; ++ni) {
        acc[0][ni] = wmma_bf16(a0, bfr[ni], acc[0][ni]);
        acc[1][ni] = wmma_bf16(a1, bfr[ni], acc[1][ni]);
      }
    }
  }
  const int rbase = mBase + wm * 32 + 8 * (lane >> 4);
  const int cbase = nBase + wn * 64 + (lane & 15);
#pragma unroll
  for (int mi = 0; mi < 2; ++mi)
#pragma unroll
    for (int ni = 0; ni < 4; ++ni)
#pragma unroll
      for (int v = 0; v < 8; ++v)
        C[(size_t)(rbase + mi * 16 + v) * N + cbase + ni * 16] =
            (TOUT)acc[mi][ni][v];
}

// ---------------------------------------------------------------------------
// NeoX rotary, in-place on bf16 q [B,S,H,HD] and k [B,S,HKV,HD].
// ---------------------------------------------------------------------------
__global__ void __launch_bounds__(256)
rope_kernel(__bf16* __restrict__ q, __bf16* __restrict__ k,
            const int* __restrict__ start_pos) {
  const int qtot = B_ * S_ * H_ * (HD_ / 2);
  const int ktot = B_ * S_ * HKV_ * (HD_ / 2);
  int idx = blockIdx.x * blockDim.x + threadIdx.x;
  if (idx >= qtot + ktot) return;
  const bool isq = idx < qtot;
  if (!isq) idx -= qtot;
  const int nh = isq ? H_ : HKV_;
  const int d = idx & 63; idx >>= 6;
  const int h = idx % nh; idx /= nh;
  const int s = idx % S_;
  const int b = idx / S_;
  const float pos = (float)(start_pos[b] + s);
  const float ang = pos * __expf(-(2.0f * (float)d / (float)HD_) * 9.210340371976184f);
  float sn, cs;
  __sincosf(ang, &sn, &cs);
  __bf16* base = isq ? (q + ((size_t)(b * S_ + s) * H_ + h) * HD_)
                     : (k + ((size_t)(b * S_ + s) * HKV_ + h) * HD_);
  const float a  = (float)base[d];
  const float bb = (float)base[d + 64];
  base[d]      = (__bf16)(a * cs - bb * sn);
  base[d + 64] = (__bf16)(bb * cs + a * sn);
}

// ---------------------------------------------------------------------------
// Flash attention, causal, GQA. One block = (b, h, 64 query rows), 8 waves.
// K-block = 64, online softmax, bf16 WMMA; K tile staged via async DMA,
// V transposed into LDS. Scale 1/sqrt(HD) applied during the softmax scan.
// ---------------------------------------------------------------------------
__global__ void __launch_bounds__(256)
attn_kernel(const __bf16* __restrict__ qg, const __bf16* __restrict__ kg,
            const __bf16* __restrict__ vg, __bf16* __restrict__ og) {
  __shared__ __attribute__((aligned(16))) __bf16 Ks[64 * 128];  // [krow][d]
  __shared__ __attribute__((aligned(16))) __bf16 Vt[128 * 64];  // [d][krow]
  __shared__ __attribute__((aligned(16))) float  Sf[64 * 64];   // raw scores
  __shared__ __attribute__((aligned(16))) __bf16 Pb[64 * 64];   // probs bf16
  __shared__ float frow[64];
  __shared__ float lrow[64];

  const int tid  = threadIdx.x;
  const int lane = tid & 31;
  const int wave = tid >> 5;
  const int ms = wave & 3;
  const int nhalf = wave >> 2;
  const int qb = blockIdx.x * 64;
  const int h  = blockIdx.y;
  const int b  = blockIdx.z;
  const int hk = h / G_;
  const float scale = 0.08838834764831845f;  // 1/sqrt(128)

  // Cache Q fragments (A-matrix) in registers: 4 d-ksteps, raw bf16 copies.
  bf16x16 qfrag[4];
  {
    const int row = qb + ms * 16 + (lane & 15);
    const __bf16* qrow = qg + ((size_t)(b * S_ + row) * H_ + h) * HD_;
#pragma unroll
    for (int kk = 0; kk < 4; ++kk) {
      const int d0 = kk * 32 + 8 * (lane >> 4);
      FragU u;
      u.h[0] = *(const bf16x8*)(qrow + d0);
      u.h[1] = *(const bf16x8*)(qrow + d0 + 16);
      qfrag[kk] = u.v;
    }
  }

  float m_run = -1e30f, l_run = 0.0f;  // live in threads tid<64
  f32x8 oacc[4];
#pragma unroll
  for (int i = 0; i < 4; ++i) oacc[i] = (f32x8)0.0f;

  const int sr  = tid >> 2;         // staged k-row 0..63
  const int sdb = (tid & 3) * 32;   // d base
  const unsigned laK = lds_addr(Ks + sr * 128 + sdb);
  const __bf16* gK = kg + ((size_t)(b * S_ + sr) * HKV_ + hk) * HD_ + sdb;
  const __bf16* gV = vg + ((size_t)(b * S_ + sr) * HKV_ + hk) * HD_ + sdb;

  for (int kb = 0; kb <= qb; kb += 64) {
    const size_t roff = (size_t)kb * HKV_ * HD_;
    __syncthreads();
    stage_row64B(laK, gK + roff);
    {
#pragma unroll
      for (int i = 0; i < 32; i += 8) {
        bf16x8 t = *(const bf16x8*)(gV + roff + i);
#pragma unroll
        for (int j = 0; j < 8; ++j) Vt[(sdb + i + j) * 64 + sr] = t[j];
      }
    }
    wait_async();
    __syncthreads();
    // Scores: raw Q @ K^T -> Sf.
#pragma unroll
    for (int nj = 0; nj < 2; ++nj) {
      const int nt = nhalf * 2 + nj;
      f32x8 sacc = (f32x8)0.0f;
#pragma unroll
      for (int kk = 0; kk < 4; ++kk) {
        bf16x16 bfr = frag_b(Ks, 128, nt * 16 + (lane & 15), kk * 32, lane);
        sacc = wmma_bf16(qfrag[kk], bfr, sacc);
      }
      const int r0 = ms * 16 + 8 * (lane >> 4);
      const int c  = nt * 16 + (lane & 15);
#pragma unroll
      for (int v = 0; v < 8; ++v) Sf[(r0 + v) * 64 + c] = sacc[v];
    }
    __syncthreads();
    // Online softmax (scale applied here): thread r<64 owns row r.
    if (tid < 64) {
      const int r = tid;
      int kmax = qb + r - kb + 1;
      if (kmax > 64) kmax = 64;
      float mx = m_run;
      for (int c = 0; c < kmax; ++c) mx = fmaxf(mx, Sf[r * 64 + c] * scale);
      const float corr = __expf(m_run - mx);
      float sum = 0.0f;
      for (int c = 0; c < 64; ++c) {
        const float p =
            (c < kmax) ? __expf(Sf[r * 64 + c] * scale - mx) : 0.0f;
        Pb[r * 64 + c] = (__bf16)p;
        sum += p;
      }
      l_run = l_run * corr + sum;
      m_run = mx;
      frow[r] = corr;
    }
    __syncthreads();
    // Rescale O, accumulate P @ V.
    float fv[8];
#pragma unroll
    for (int v = 0; v < 8; ++v) fv[v] = frow[ms * 16 + v + 8 * (lane >> 4)];
#pragma unroll
    for (int dt = 0; dt < 4; ++dt)
#pragma unroll
      for (int v = 0; v < 8; ++v) oacc[dt][v] *= fv[v];
#pragma unroll
    for (int kk = 0; kk < 2; ++kk) {
      bf16x16 pa = frag_a(Pb, 64, ms * 16 + (lane & 15), kk * 32, lane);
#pragma unroll
      for (int dt = 0; dt < 4; ++dt) {
        bf16x16 bv = frag_b(Vt, 64, (nhalf * 4 + dt) * 16 + (lane & 15),
                            kk * 32, lane);
        oacc[dt] = wmma_bf16(pa, bv, oacc[dt]);
      }
    }
  }
  if (tid < 64) lrow[tid] = 1.0f / l_run;
  __syncthreads();
  float inv[8];
#pragma unroll
  for (int v = 0; v < 8; ++v) inv[v] = lrow[ms * 16 + v + 8 * (lane >> 4)];
#pragma unroll
  for (int dt = 0; dt < 4; ++dt) {
    const int col = (nhalf * 4 + dt) * 16 + (lane & 15);
#pragma unroll
    for (int v = 0; v < 8; ++v) {
      const int row = qb + ms * 16 + v + 8 * (lane >> 4);
      og[((size_t)(b * S_ + row) * H_ + h) * HD_ + col] =
          (__bf16)(oacc[dt][v] * inv[v]);
    }
  }
}

// ---------------------------------------------------------------------------
// Launch. Scratch (bf16 elements):
//   xb[2048*4096] wqT[4096*4096] wkT[1024*4096] wvT[1024*4096] woT[4096*4096]
//   q[2048*4096] k[2048*1024] v[2048*1024] a[2048*4096]   (~143 MB total)
// ---------------------------------------------------------------------------
extern "C" void kernel_launch(void* const* d_in, const int* in_sizes, int n_in,
                              void* d_out, int out_size, void* d_ws, size_t ws_size,
                              hipStream_t stream) {
  const float* x  = (const float*)d_in[0];
  const float* wq = (const float*)d_in[2];
  const float* wk = (const float*)d_in[3];
  const float* wv = (const float*)d_in[4];
  const float* wo = (const float*)d_in[5];
  const int* start_pos = (const int*)d_in[10];
  float* out = (float*)d_out;

  const int M = B_ * S_;  // 2048
  __bf16* xb  = (__bf16*)d_ws;
  __bf16* wqT = xb  + (size_t)M * HID_;
  __bf16* wkT = wqT + (size_t)HID_ * (H_ * HD_);
  __bf16* wvT = wkT + (size_t)HID_ * (HKV_ * HD_);
  __bf16* woT = wvT + (size_t)HID_ * (HKV_ * HD_);
  __bf16* qb  = woT + (size_t)(H_ * HD_) * HID_;
  __bf16* kb  = qb  + (size_t)M * (H_ * HD_);
  __bf16* vb  = kb  + (size_t)M * (HKV_ * HD_);
  __bf16* ab  = vb  + (size_t)M * (HKV_ * HD_);

  // One-time conversions: x -> bf16; weights -> bf16 transposed [N,K].
  {
    int n = M * HID_;
    cast_f32_bf16_kernel<<<(n + 255) / 256, 256, 0, stream>>>(x, xb, n);
  }
  transpose_f32_bf16_kernel<<<(HID_ * H_ * HD_ + 255) / 256, 256, 0, stream>>>(
      wq, wqT, HID_, H_ * HD_);
  transpose_f32_bf16_kernel<<<(HID_ * HKV_ * HD_ + 255) / 256, 256, 0, stream>>>(
      wk, wkT, HID_, HKV_ * HD_);
  transpose_f32_bf16_kernel<<<(HID_ * HKV_ * HD_ + 255) / 256, 256, 0, stream>>>(
      wv, wvT, HID_, HKV_ * HD_);
  transpose_f32_bf16_kernel<<<(H_ * HD_ * HID_ + 255) / 256, 256, 0, stream>>>(
      wo, woT, H_ * HD_, HID_);

  // QKV projections (async-staged bf16 WMMA GEMMs, bf16 outputs).
  gemm_bf16_async_kernel<__bf16><<<dim3(H_ * HD_ / 128, M / 128), 256, 0, stream>>>(
      xb, wqT, qb, M, H_ * HD_, HID_);
  gemm_bf16_async_kernel<__bf16><<<dim3(HKV_ * HD_ / 128, M / 128), 256, 0, stream>>>(
      xb, wkT, kb, M, HKV_ * HD_, HID_);
  gemm_bf16_async_kernel<__bf16><<<dim3(HKV_ * HD_ / 128, M / 128), 256, 0, stream>>>(
      xb, wvT, vb, M, HKV_ * HD_, HID_);

  // Rotary on q and k (bf16 in-place).
  const int rtot = B_ * S_ * (H_ + HKV_) * (HD_ / 2);
  rope_kernel<<<(rtot + 255) / 256, 256, 0, stream>>>(qb, kb, start_pos);

  // Causal GQA flash attention (bf16 out).
  attn_kernel<<<dim3(S_ / 64, H_, B_), 256, 0, stream>>>(qb, kb, vb, ab);

  // Output projection (f32 out).
  gemm_bf16_async_kernel<float><<<dim3(HID_ / 128, M / 128), 256, 0, stream>>>(
      ab, woT, out, M, HID_, H_ * HD_);
}
